// FutureLSTM_90855738180292
// MI455X (gfx1250) — matrix operand order repaired and checked by reading, not compiled, not run on hardware
//
#include <hip/hip_runtime.h>
#include <hip/hip_bf16.h>

// ---------------------------------------------------------------------------
// FutureLSTM on MI455X (gfx1250): bf16 WMMA, BM=32 rows per block.
// 512-thread blocks (16 waves): 2 gate jobs/wave so persistent cell-state c
// fits in 32 VGPRs and total pressure stays < 256 (no s_set_vgpr_msb, no
// spills). Activations (out | h double-buffer) live in 128.5 KB LDS across
// all 8 decode steps -> 2 blocks per 320KB WGP -> 8 waves/SIMD.
// Weights pre-packed into WMMA B-fragment lane order -> coalesced 1KB wave
// reads from L2. K loops rolled (unroll 1).
// ---------------------------------------------------------------------------

typedef __attribute__((ext_vector_type(16))) __bf16 v16bf;
typedef __attribute__((ext_vector_type(8)))  float  v8f;

#define BM     32        // rows per block (2 WMMA M-tiles)
#define HDIM   512
#define FDIM   1024
#define NROWS  8192      // B*E
#define ACT_LD 2056      // [out(1024) | h0(512) | h1(512)] + 8 pad (bank de-conflict)
#define H_OFF  1024
#define NTHR   512       // 16 waves

__device__ __forceinline__ unsigned short f32_to_bf16_raw(float f) {
    union { float f; unsigned u; } v; v.f = f;
    unsigned x = v.u;
    x += 0x7FFFu + ((x >> 16) & 1u);          // round-to-nearest-even
    return (unsigned short)(x >> 16);
}

__device__ __forceinline__ float sigmoidf_(float x) {
    return 1.0f / (1.0f + __expf(-x));
}

// A fragment (16x32 bf16): p = row/lane base; two contiguous 16B LDS loads.
__device__ __forceinline__ v16bf fragA(const unsigned short* p) {
    union { uint4 u[2]; v16bf v; } r;
    r.u[0] = *(const uint4*)(p);
    r.u[1] = *(const uint4*)(p + 16);
    return r.v;
}

// B fragment from pre-packed weights: 16 contiguous bf16 at p (per lane).
__device__ __forceinline__ v16bf fragB(const unsigned short* __restrict__ p) {
    union { uint4 u[2]; v16bf v; } r;
    r.u[0] = *(const uint4*)(p);
    r.u[1] = *(const uint4*)(p + 8);
    return r.v;
}

__device__ __forceinline__ v8f wmma_bf16(v16bf a, v16bf b, v8f c) {
    return __builtin_amdgcn_wmma_f32_16x16x32_bf16(false, a, false, b,
                                                   (short)0, c, false, false);
}

// ---------------------------------------------------------------------------
// Prep kernels
// ---------------------------------------------------------------------------
__global__ void cvt_f32_to_bf16(const float* __restrict__ in,
                                unsigned short* __restrict__ out, int n) {
    int i = blockIdx.x * blockDim.x + threadIdx.x;
    if (i < n) out[i] = f32_to_bf16_raw(in[i]);
}

// Pack W[N][K] (f32 row-major) into WMMA B-fragment order:
// P[nt][kt][lane][j] = bf16( W[nt*16 + lane%16][kt*32 + (j/8)*16 + (lane/16)*8 + j%8] )
__global__ void pack_weight_bf16(const float* __restrict__ W,
                                 unsigned short* __restrict__ P, int N, int K) {
    int idx = blockIdx.x * blockDim.x + threadIdx.x;
    if (idx >= N * K) return;
    const int ktiles = K >> 5;
    int j    = idx & 15;
    int lane = (idx >> 4) & 31;
    int kt   = (idx >> 9) % ktiles;
    int nt   = idx / (ktiles << 9);
    int n = nt * 16 + (lane & 15);
    int k = kt * 32 + ((j >> 3) << 4) + ((lane >> 4) << 3) + (j & 7);
    P[idx] = f32_to_bf16_raw(W[(size_t)n * K + k]);
}

__global__ void fuse_bias(const float* __restrict__ bih,
                          const float* __restrict__ bhh,
                          float* __restrict__ bias, int n) {
    int i = blockIdx.x * blockDim.x + threadIdx.x;
    if (i < n) bias[i] = bih[i] + bhh[i];
}

// ---------------------------------------------------------------------------
// One gate job: (32 rows x 16 H-cols) -> gates GEMM + pointwise LSTM cell.
// c0/c1 are the persistent per-lane cell-state registers for this tile pair.
// ---------------------------------------------------------------------------
__device__ __forceinline__ void gates_job(
    unsigned short* act, int aoff0, int aoff1, int hrd, int hwr,
    const unsigned short* __restrict__ Pih,
    const unsigned short* __restrict__ Phh,
    const float* __restrict__ bias,
    int hj, int lane, v8f& c0, v8f& c1)
{
    const int lm  = lane & 15;
    const int sel = lane >> 4;

    // packed B bases: Wih tile stride 32*512, Whh tile stride 16*512
    const unsigned short* bI = Pih + (size_t)(hj)       * 16384 + lane * 16;
    const unsigned short* bF = Pih + (size_t)(32 + hj)  * 16384 + lane * 16;
    const unsigned short* bG = Pih + (size_t)(64 + hj)  * 16384 + lane * 16;
    const unsigned short* bO = Pih + (size_t)(96 + hj)  * 16384 + lane * 16;
    const unsigned short* cI = Phh + (size_t)(hj)       * 8192  + lane * 16;
    const unsigned short* cF = Phh + (size_t)(32 + hj)  * 8192  + lane * 16;
    const unsigned short* cG = Phh + (size_t)(64 + hj)  * 8192  + lane * 16;
    const unsigned short* cO = Phh + (size_t)(96 + hj)  * 8192  + lane * 16;

    v8f ai0 = {}, af0 = {}, ag0 = {}, ao0 = {};
    v8f ai1 = {}, af1 = {}, ag1 = {}, ao1 = {};

    // out[prev] @ W_ih^T : K = 1024 (32 kt)
    #pragma unroll 1
    for (int kt = 0; kt < 32; ++kt) {
        v16bf A0 = fragA(act + aoff0 + kt * 32);
        v16bf A1 = fragA(act + aoff1 + kt * 32);
        v16bf Bi = fragB(bI + kt * 512);
        v16bf Bf = fragB(bF + kt * 512);
        v16bf Bg = fragB(bG + kt * 512);
        v16bf Bo = fragB(bO + kt * 512);
        ai0 = wmma_bf16(A0, Bi, ai0);  ai1 = wmma_bf16(A1, Bi, ai1);
        af0 = wmma_bf16(A0, Bf, af0);  af1 = wmma_bf16(A1, Bf, af1);
        ag0 = wmma_bf16(A0, Bg, ag0);  ag1 = wmma_bf16(A1, Bg, ag1);
        ao0 = wmma_bf16(A0, Bo, ao0);  ao1 = wmma_bf16(A1, Bo, ao1);
    }
    // h[prev] @ W_hh^T : K = 512 (16 kt)
    #pragma unroll 1
    for (int kt = 0; kt < 16; ++kt) {
        v16bf A0 = fragA(act + aoff0 + hrd + kt * 32);
        v16bf A1 = fragA(act + aoff1 + hrd + kt * 32);
        v16bf Bi = fragB(cI + kt * 512);
        v16bf Bf = fragB(cF + kt * 512);
        v16bf Bg = fragB(cG + kt * 512);
        v16bf Bo = fragB(cO + kt * 512);
        ai0 = wmma_bf16(A0, Bi, ai0);  ai1 = wmma_bf16(A1, Bi, ai1);
        af0 = wmma_bf16(A0, Bf, af0);  af1 = wmma_bf16(A1, Bf, af1);
        ag0 = wmma_bf16(A0, Bg, ag0);  ag1 = wmma_bf16(A1, Bg, ag1);
        ao0 = wmma_bf16(A0, Bo, ao0);  ao1 = wmma_bf16(A1, Bo, ao1);
    }

    // pointwise LSTM cell (c in registers)
    const int n   = hj * 16 + lm;
    const float bIv = bias[n];
    const float bFv = bias[HDIM + n];
    const float bGv = bias[2 * HDIM + n];
    const float bOv = bias[3 * HDIM + n];

    #pragma unroll
    for (int r = 0; r < 8; ++r) {             // M-tile 0: rows 0..15
        const int m = sel * 8 + r;
        const float iv = sigmoidf_(ai0[r] + bIv);
        const float fv = sigmoidf_(af0[r] + bFv);
        const float gv = tanhf(ag0[r] + bGv);
        const float ov = sigmoidf_(ao0[r] + bOv);
        const float cn = fv * c0[r] + iv * gv;
        c0[r] = cn;
        act[m * ACT_LD + hwr + n] = f32_to_bf16_raw(ov * tanhf(cn));
    }
    #pragma unroll
    for (int r = 0; r < 8; ++r) {             // M-tile 1: rows 16..31
        const int m = 16 + sel * 8 + r;
        const float iv = sigmoidf_(ai1[r] + bIv);
        const float fv = sigmoidf_(af1[r] + bFv);
        const float gv = tanhf(ag1[r] + bGv);
        const float ov = sigmoidf_(ao1[r] + bOv);
        const float cn = fv * c1[r] + iv * gv;
        c1[r] = cn;
        act[m * ACT_LD + hwr + n] = f32_to_bf16_raw(ov * tanhf(cn));
    }
}

// ---------------------------------------------------------------------------
// Main persistent LSTM kernel: one block = 32 rows, 16 waves, dynamic LDS.
// ---------------------------------------------------------------------------
__global__ __launch_bounds__(NTHR) void lstm_wmma_kernel(
    const unsigned short* __restrict__ X,     // [8192,1024] bf16
    const unsigned short* __restrict__ Pih,   // packed [128][32][512]
    const unsigned short* __restrict__ Phh,   // packed [128][16][512]
    const unsigned short* __restrict__ Pcls,  // packed [ 64][16][512]
    const float* __restrict__ bias,           // [2048] b_ih+b_hh
    const float* __restrict__ bcls,           // [1024]
    float* __restrict__ out,                  // [8192, D, 1024]
    const int* __restrict__ dec)              // scalar D
{
    extern __shared__ __align__(16) char smem[];
    unsigned short* act = (unsigned short*)smem;               // [32][ACT_LD]

    const int tid  = threadIdx.x;
    const int lane = tid & 31;
    const int wave = tid >> 5;                 // 0..15
    const int lm   = lane & 15;
    const int sel  = lane >> 4;                // 0/1
    const int row0 = blockIdx.x * BM;
    const int D    = dec[0];

    // ---- init: out cols = x slab, h regions = 0 ----
    {
        for (int i = tid; i < BM * 128; i += NTHR) {          // out: 128 uint4/row
            int r = i >> 7, q = i & 127;
            ((uint4*)(act + r * ACT_LD))[q] =
                ((const uint4*)(X + (size_t)(row0 + r) * FDIM))[q];
        }
        uint4 z; z.x = z.y = z.z = z.w = 0u;
        for (int i = tid; i < BM * 128; i += NTHR) {          // h0|h1: 128 uint4/row
            int r = i >> 7, q = i & 127;
            ((uint4*)(act + r * ACT_LD + H_OFF))[q] = z;
        }
    }
    __syncthreads();

    // per-lane A bases (row-major LDS, two M-tiles)
    const int aoff0 = lm * ACT_LD + sel * 8;
    const int aoff1 = (16 + lm) * ACT_LD + sel * 8;

    // persistent cell state in registers: 2 jobs x 2 M-tiles = 32 VGPRs
    v8f c00 = {}, c01 = {}, c10 = {}, c11 = {};

    int p = 0;
    for (int s = 0; s < D; ++s) {
        const int hrd = H_OFF + p * HDIM;          // read  h[p]
        const int hwr = H_OFF + (p ^ 1) * HDIM;    // write h[p^1]

        // ===== gates + pointwise LSTM: 2 jobs per wave (unrolled for c regs) ==
        gates_job(act, aoff0, aoff1, hrd, hwr, Pih, Phh, bias, wave * 2 + 0, lane, c00, c01);
        gates_job(act, aoff0, aoff1, hrd, hwr, Pih, Phh, bias, wave * 2 + 1, lane, c10, c11);
        __syncthreads();   // h[p^1] complete; all out-region reads done

        // ============ classifier: out = h_new @ W_cls^T + b ============
        #pragma unroll 1
        for (int j = 0; j < 4; ++j) {
            const int nt = wave * 4 + j;           // 64 N-tiles / 16 waves
            const unsigned short* bC = Pcls + (size_t)nt * 8192 + lane * 16;
            v8f d0 = {}, d1 = {};
            #pragma unroll 1
            for (int kt = 0; kt < 16; ++kt) {
                v16bf A0 = fragA(act + aoff0 + hwr + kt * 32);
                v16bf A1 = fragA(act + aoff1 + hwr + kt * 32);
                v16bf B  = fragB(bC + kt * 512);
                d0 = wmma_bf16(A0, B, d0);
                d1 = wmma_bf16(A1, B, d1);
            }
            const int n = nt * 16 + lm;
            const float bC0 = bcls[n];
            #pragma unroll
            for (int r = 0; r < 8; ++r) {          // M-tile 0
                const int m = sel * 8 + r;
                const float v = d0[r] + bC0;
                act[m * ACT_LD + n] = f32_to_bf16_raw(v);
                out[((size_t)(row0 + m) * D + s) * FDIM + n] = v;
            }
            #pragma unroll
            for (int r = 0; r < 8; ++r) {          // M-tile 1
                const int m = 16 + sel * 8 + r;
                const float v = d1[r] + bC0;
                act[m * ACT_LD + n] = f32_to_bf16_raw(v);
                out[((size_t)(row0 + m) * D + s) * FDIM + n] = v;
            }
        }
        p ^= 1;
        __syncthreads();   // new out/h visible before next step
    }
}

// ---------------------------------------------------------------------------
// Launcher
// ---------------------------------------------------------------------------
extern "C" void kernel_launch(void* const* d_in, const int* in_sizes, int n_in,
                              void* d_out, int out_size, void* d_ws, size_t ws_size,
                              hipStream_t stream) {
    (void)in_sizes; (void)n_in; (void)out_size; (void)ws_size;

    const float* x     = (const float*)d_in[0];
    const float* W_ih  = (const float*)d_in[1];
    const float* W_hh  = (const float*)d_in[2];
    const float* b_ih  = (const float*)d_in[3];
    const float* b_hh  = (const float*)d_in[4];
    const float* W_cls = (const float*)d_in[5];
    const float* b_cls = (const float*)d_in[6];
    const int*   dec   = (const int*)d_in[7];

    const int nX   = NROWS * FDIM;        // 8388608
    const int nWih = 4 * HDIM * FDIM;     // 2097152
    const int nWhh = 4 * HDIM * HDIM;     // 1048576
    const int nWcl = FDIM * HDIM;         // 524288
    const int nB   = 4 * HDIM;            // 2048

    char* ws = (char*)d_ws;
    unsigned short* Xb   = (unsigned short*)ws;   ws += (size_t)nX   * 2;
    unsigned short* Pih  = (unsigned short*)ws;   ws += (size_t)nWih * 2;
    unsigned short* Phh  = (unsigned short*)ws;   ws += (size_t)nWhh * 2;
    unsigned short* Pcls = (unsigned short*)ws;   ws += (size_t)nWcl * 2;
    float*          biasF = (float*)ws;

    cvt_f32_to_bf16<<<(nX + 255) / 256, 256, 0, stream>>>(x, Xb, nX);
    pack_weight_bf16<<<(nWih + 255) / 256, 256, 0, stream>>>(W_ih,  Pih,  4 * HDIM, FDIM);
    pack_weight_bf16<<<(nWhh + 255) / 256, 256, 0, stream>>>(W_hh,  Phh,  4 * HDIM, HDIM);
    pack_weight_bf16<<<(nWcl + 255) / 256, 256, 0, stream>>>(W_cls, Pcls, FDIM,     HDIM);
    fuse_bias<<<(nB + 255) / 256, 256, 0, stream>>>(b_ih, b_hh, biasF, nB);

    const size_t smem = (size_t)BM * ACT_LD * 2;   // 128.5 KB (act only)
    lstm_wmma_kernel<<<NROWS / BM, NTHR, smem, stream>>>(
        Xb, Pih, Phh, Pcls, biasF, b_cls, (float*)d_out, dec);
}
